// WindowShiftedMDTA_14989435863613
// MI455X (gfx1250) — matrix operand order repaired and checked by reading, not compiled, not run on hardware
//
#include <hip/hip_runtime.h>
#include <hip/hip_bf16.h>

typedef __attribute__((ext_vector_type(16))) _Float16 v16h;
typedef __attribute__((ext_vector_type(8)))  _Float16 v8h;
typedef __attribute__((ext_vector_type(8)))  float    v8f;
typedef __attribute__((ext_vector_type(4)))  float    v4f;

#define LDX 264          // halfs: [64 tokens][256 ch] padded (bank spread, 16B-aligned rows)
#define LDV 72           // halfs: Vp [256 ch][64 tok] padded
#define LDA 72           // halfs: attn [64][64] padded
#define AHEAD (64*LDA)   // halfs per attention head

union AFrag { v16h v; v8h h[2]; };
union DFrag { v8f v; v4f q[2]; };

// A-matrix fragment, 16(M)x32(K) f16, source row-major [M][K], ld in halfs.
// Layout: lane<16 -> M=lane, K=0..15; lane>=16 -> M=lane-16, K=16..31.
__device__ __forceinline__ v16h loadA16x32(const _Float16* base, int ld) {
  const int lane = threadIdx.x & 31;
  const _Float16* p = base + (lane & 15) * ld + (lane >> 4) * 16;
  AFrag u;
  u.h[0] = *(const v8h*)(p);
  u.h[1] = *(const v8h*)(p + 8);
  return u.v;
}

// B-matrix fragment, 32(K)x16(N) f16, source stored TRANSPOSED row-major [N][K], ld in halfs.
// Layout: lane<16 -> N=lane, K=0..15; lane>=16 -> N=lane-16, K=16..31 (2 halfs per VGPR).
__device__ __forceinline__ v16h loadB32x16(const _Float16* base, int ld) {
  const int lane = threadIdx.x & 31;
  const _Float16* p = base + (lane & 15) * ld + (lane >> 4) * 16;
  AFrag u;
  u.h[0] = *(const v8h*)(p);
  u.h[1] = *(const v8h*)(p + 8);
  return u.v;
}

// D tile element (M=8*hi+r, N=lane&15); store the 8 per-lane rows as 8 contiguous f16.
__device__ __forceinline__ void storeD_f16(_Float16* p, v8f d) {
  v8h o;
  #pragma unroll
  for (int r = 0; r < 8; ++r) o[r] = (_Float16)d[r];
  *(v8h*)p = o;
}

// Weight f32 -> f16 (row-major 256x256 each), once per launch into workspace.
__global__ void cvt_w_kernel(const float* __restrict__ a, const float* __restrict__ b,
                             const float* __restrict__ c, const float* __restrict__ d,
                             _Float16* __restrict__ o) {
  int i = blockIdx.x * blockDim.x + threadIdx.x;
  if (i < 65536) {
    o[i]          = (_Float16)a[i];
    o[i + 65536]  = (_Float16)b[i];
    o[i + 131072] = (_Float16)c[i];
    o[i + 196608] = (_Float16)d[i];
  }
}

__global__ __launch_bounds__(256)
void mdta_win_kernel(const float* __restrict__ xq_g, const float* __restrict__ xkv_g,
                     const float* __restrict__ temp, const _Float16* __restrict__ wf,
                     float* __restrict__ out, int nwin_per_img) {
  extern __shared__ _Float16 smem[];
  // Region layout (halfs). attn overlays Xq/Xkv (dead after phase 2);
  // O overlays Qp (dead after phase 3). Peak = 89088 halfs = 178176 B.
  _Float16* Xq   = smem;                       // [64][LDX]
  _Float16* Xkv  = smem + 64 * LDX;            // [64][LDX]
  _Float16* attn = smem;                       // [8][64][LDA]
  _Float16* Qp   = smem + 36864;               // [64][LDX]
  _Float16* Op   = Qp;                         // [64][LDX]
  _Float16* Kp   = smem + 36864 + 64 * LDX;    // [64][LDX]
  _Float16* Vp   = smem + 36864 + 2 * 64 * LDX;// [256][LDV]

  const _Float16* wq = wf;
  const _Float16* wk = wf + 65536;
  const _Float16* wv = wf + 131072;
  const _Float16* wo = wf + 196608;

  const int w  = blockIdx.x;
  const int b  = w / nwin_per_img;
  const int rw = w % nwin_per_img;
  const int wh = rw / 24, ww = rw % 24;        // 24 = 192/8 windows per row
  const int tid  = threadIdx.x;
  const int wvid = tid >> 5;
  const int lane = tid & 31;
  const int ln15 = lane & 15;
  const int hi   = lane >> 4;

  const float* xqb = xq_g  + (size_t)b * 256 * 36864;
  const float* xkb = xkv_g + (size_t)b * 256 * 36864;

  // ---- Phase 1: gather window, f32 -> f16, token-major into LDS ----
  #pragma unroll 4
  for (int idx = tid; idx < 16384; idx += 256) {
    int token = idx & 63, c = idx >> 6;
    int i = token >> 3, j = token & 7;
    size_t g = (size_t)c * 36864 + (size_t)(wh * 8 + i) * 192 + (size_t)(ww * 8 + j);
    Xq[token * LDX + c]  = (_Float16)xqb[g];
    Xkv[token * LDX + c] = (_Float16)xkb[g];
  }
  __syncthreads();

  // ---- Phase 2: Q = Wq*X, K = Wk*X (M=c_out,N=token); V computed as X^T*Wv^T (M=token,N=c_out)
  for (int t = wvid; t < 192; t += 8) {
    int g = t >> 6, i = t & 63;
    v8f acc = {};
    if (g < 2) {
      const _Float16* wm = g ? wk : wq;
      const _Float16* xb = g ? Xkv : Xq;
      _Float16* dst      = g ? Kp : Qp;
      int ct = i >> 2, tt = i & 3;
      #pragma unroll
      for (int k = 0; k < 8; ++k) {
        v16h a  = loadA16x32(wm + ct * 16 * 256 + k * 32, 256);
        v16h bf = loadB32x16(xb + tt * 16 * LDX + k * 32, LDX);
        acc = __builtin_amdgcn_wmma_f32_16x16x32_f16(false, a, false, bf, (short)0, acc, false, false);
      }
      // D (c_out, token) -> dst[token][c_out]: 8 contiguous halfs per lane
      storeD_f16(dst + (tt * 16 + ln15) * LDX + ct * 16 + hi * 8, acc);
    } else {
      int tt = i >> 4, ct = i & 15;
      #pragma unroll
      for (int k = 0; k < 8; ++k) {
        v16h a  = loadA16x32(Xkv + tt * 16 * LDX + k * 32, LDX);
        v16h bf = loadB32x16(wv + ct * 16 * 256 + k * 32, 256);
        acc = __builtin_amdgcn_wmma_f32_16x16x32_f16(false, a, false, bf, (short)0, acc, false, false);
      }
      // D (token, c_out) -> Vp[c_out][token]: 8 contiguous halfs per lane
      storeD_f16(Vp + (ct * 16 + ln15) * LDV + tt * 16 + hi * 8, acc);
    }
  }
  __syncthreads();

  // ---- Phase 3: per head (wave = head): S = K^T Q, temperature, softmax over m (= M dim) ----
  {
    const int h = wvid;
    const float tpr = temp[h];
    v16h bq[4];
    #pragma unroll
    for (int tn = 0; tn < 4; ++tn) bq[tn] = loadB32x16(Qp + tn * 16 * LDX + h * 32, LDX);
    v8f acc[4][4];
    #pragma unroll
    for (int tm = 0; tm < 4; ++tm) {
      v16h a = loadA16x32(Kp + tm * 16 * LDX + h * 32, LDX);
      #pragma unroll
      for (int tn = 0; tn < 4; ++tn) {
        v8f z = {};
        acc[tm][tn] = __builtin_amdgcn_wmma_f32_16x16x32_f16(false, a, false, bq[tn], (short)0, z, false, false);
      }
    }
    #pragma unroll
    for (int tn = 0; tn < 4; ++tn) {
      float mx = -3.4e38f;
      #pragma unroll
      for (int tm = 0; tm < 4; ++tm)
        #pragma unroll
        for (int r = 0; r < 8; ++r) {
          float s = acc[tm][tn][r] * tpr;
          acc[tm][tn][r] = s;
          mx = fmaxf(mx, s);
        }
      mx = fmaxf(mx, __shfl_xor(mx, 16, 32));   // fold lanes 0-15 with 16-31 (same n)
      float sum = 0.f;
      #pragma unroll
      for (int tm = 0; tm < 4; ++tm)
        #pragma unroll
        for (int r = 0; r < 8; ++r) {
          float e = __expf(acc[tm][tn][r] - mx);
          acc[tm][tn][r] = e;
          sum += e;
        }
      sum += __shfl_xor(sum, 16, 32);
      float inv = __builtin_amdgcn_rcpf(sum);
      int n = tn * 16 + ln15;
      #pragma unroll
      for (int tm = 0; tm < 4; ++tm) {
        v8h o;
        #pragma unroll
        for (int r = 0; r < 8; ++r) o[r] = (_Float16)(acc[tm][tn][r] * inv);
        *(v8h*)(attn + h * AHEAD + n * LDA + tm * 16 + hi * 8) = o;  // attn[n][m]
      }
    }
  }
  __syncthreads();

  // ---- Phase 4: O^T per head: M=d, N=n; A = Vp (d-major), B = attn ([n][m]) ----
  {
    const int h = wvid;
    #pragma unroll
    for (int dt = 0; dt < 2; ++dt)
      #pragma unroll
      for (int tn = 0; tn < 4; ++tn) {
        v8f acc = {};
        #pragma unroll
        for (int k = 0; k < 2; ++k) {
          v16h a  = loadA16x32(Vp + (h * 32 + dt * 16) * LDV + k * 32, LDV);
          v16h bf = loadB32x16(attn + h * AHEAD + tn * 16 * LDA + k * 32, LDA);
          acc = __builtin_amdgcn_wmma_f32_16x16x32_f16(false, a, false, bf, (short)0, acc, false, false);
        }
        // D (d, n) -> Op[n][h*32 + d]: contiguous
        storeD_f16(Op + (tn * 16 + ln15) * LDX + h * 32 + dt * 16 + hi * 8, acc);
      }
  }
  __syncthreads();

  // ---- Phase 5: Out = O * Wo^T (M=token, N=c_out); fused window-merge store ----
  for (int t = wvid; t < 64; t += 8) {
    int tm = t >> 4, ct = t & 15;
    v8f acc = {};
    #pragma unroll
    for (int k = 0; k < 8; ++k) {
      v16h a  = loadA16x32(Op + tm * 16 * LDX + k * 32, LDX);
      v16h bf = loadB32x16(wo + ct * 16 * 256 + k * 32, 256);
      acc = __builtin_amdgcn_wmma_f32_16x16x32_f16(false, a, false, bf, (short)0, acc, false, false);
    }
    // lane: c_out fixed, tokens tm*16+8*hi+r (r=0..7) = one window row, j = 0..7 contiguous
    int co = ct * 16 + ln15;
    int i  = tm * 2 + hi;
    float* dst = out + ((size_t)b * 256 + co) * 36864 + (size_t)(wh * 8 + i) * 192 + (size_t)(ww * 8);
    DFrag d; d.v = acc;
    *(v4f*)(dst)     = d.q[0];
    *(v4f*)(dst + 4) = d.q[1];
  }
}

extern "C" void kernel_launch(void* const* d_in, const int* in_sizes, int n_in,
                              void* d_out, int out_size, void* d_ws, size_t ws_size,
                              hipStream_t stream) {
  const float* xq   = (const float*)d_in[0];
  const float* xkv  = (const float*)d_in[1];
  const float* temp = (const float*)d_in[2];
  const float* Wq   = (const float*)d_in[3];
  const float* Wk   = (const float*)d_in[4];
  const float* Wv   = (const float*)d_in[5];
  const float* Wo   = (const float*)d_in[6];
  _Float16* wf = (_Float16*)d_ws;           // 4 x 256x256 f16 = 512 KB
  float* out = (float*)d_out;

  int B = in_sizes[0] / (256 * 192 * 192);  // = 2
  int nwin_per_img = (192 / 8) * (192 / 8); // 576
  int nblocks = B * nwin_per_img;           // 1152 windows

  cvt_w_kernel<<<256, 256, 0, stream>>>(Wq, Wk, Wv, Wo, wf);

  const size_t smem_bytes = 178176;         // 174 KB dynamic LDS
  (void)hipFuncSetAttribute((const void*)mdta_win_kernel,
                            hipFuncAttributeMaxDynamicSharedMemorySize, (int)smem_bytes);
  mdta_win_kernel<<<nblocks, 256, smem_bytes, stream>>>(xq, xkv, temp, wf, out, nwin_per_img);
}
